// SchNetInteractionBlock_25701084299911
// MI455X (gfx1250) — compile-verified, matrix operand order checked by reading, not run on hardware
//
#include <hip/hip_runtime.h>
#include <hip/hip_bf16.h>

#define N_ATOMS   20000
#define N_PAIRS   640000
#define N_BASIS   128
#define N_FILTERS 128
#define N_RBF     20

#define APITCH 136   // bf16 pitch for K=128 tiles (272B/row -> 4-bank skew)
#define FPITCH 40    // bf16 pitch for K=32 (rbf padded) tiles (80B/row, 16B aligned)

typedef __bf16 bf16;
typedef __attribute__((ext_vector_type(16))) __bf16 v16bf;
typedef __attribute__((ext_vector_type(8)))  float  v8f;
typedef __attribute__((ext_vector_type(4)))  unsigned int u32x4;

// Shifted softplus: softplus(v) - log(2), numerically stable.
__device__ __forceinline__ float ssp(float v) {
    float a = fabsf(v);
    return fmaxf(v, 0.0f) + __logf(1.0f + __expf(-a)) - 0.69314718056f;
}

// Load one 16-lane-pattern WMMA fragment (A row or B column) from LDS.
// Per ISA 7.12.2 (16-bit A 16x32): elements 0..7 <-> K = kbase + 8*half + 0..7,
// elements 8..15 <-> K = kbase + 16 + 8*half + 0..7.  Two ds_load_b128 per frag.
__device__ __forceinline__ v16bf load_frag(const bf16* base, int idx, int pitch,
                                           int kbase, int half) {
    const bf16* p = base + idx * pitch + kbase + half * 8;
    v16bf r;
    ((u32x4*)&r)[0] = *(const u32x4*)(p);
    ((u32x4*)&r)[1] = *(const u32x4*)(p + 16);
    return r;
}

__device__ __forceinline__ v8f wmma_bf16(v16bf a, v16bf b, v8f c) {
    return __builtin_amdgcn_wmma_f32_16x16x32_bf16(
        /*neg_a=*/false, a, /*neg_b=*/false, b,
        /*c_mod=*/(short)0, c, /*reuse_a=*/false, /*reuse_b=*/false);
}

// CDNA5 async global->LDS copy (ASYNCcnt-tracked), 16B per lane.
// dst_lds_off = byte offset within the workgroup LDS segment (low 32 bits of
// a generic pointer to __shared__ memory); src = per-lane global address.
__device__ __forceinline__ void async_copy_b128(unsigned dst_lds_off,
                                                const void* src) {
    asm volatile("global_load_async_to_lds_b128 %0, %1, off"
                 :: "v"(dst_lds_off), "v"(src)
                 : "memory");
}

__device__ __forceinline__ void wait_async0() {
    asm volatile("s_wait_asynccnt 0x0" ::: "memory");
}

__global__ void zero_kernel(float* __restrict__ p, int n) {
    int i = blockIdx.x * blockDim.x + threadIdx.x;
    if (i < n) p[i] = 0.0f;
}

// -------------------------------------------------------------------------
// Kernel 1:  h = x @ W_in + b_in            [20000,128] x [128,128]
// Block = 256 threads (8 waves), one 16-row atom tile, wave w -> N-tile w.
// -------------------------------------------------------------------------
__global__ void gemm_h_kernel(const float* __restrict__ x,
                              const float* __restrict__ Win,
                              const float* __restrict__ bin,
                              float* __restrict__ h) {
    __shared__ __attribute__((aligned(16))) bf16 sA[16 * APITCH];
    __shared__ __attribute__((aligned(16))) bf16 sB[128 * APITCH]; // col-major [n][k]
    const int tid = threadIdx.x;
    const int mt  = blockIdx.x;                     // atom tile 0..1249

    for (int i = tid; i < 16 * 128; i += 256) {
        int r = i >> 7, k = i & 127;
        sA[r * APITCH + k] = (bf16)x[(mt * 16 + r) * N_BASIS + k];
    }
    for (int i = tid; i < 128 * 128; i += 256) {
        int k = i >> 7, n = i & 127;
        sB[n * APITCH + k] = (bf16)Win[k * N_FILTERS + n];
    }
    __syncthreads();

    const int wave = tid >> 5, lane = tid & 31;
    const int half = lane >> 4, lc = lane & 15;
    const int col  = wave * 16 + lc;

    v8f acc = {};
#pragma unroll
    for (int kc = 0; kc < 128; kc += 32) {
        v16bf a = load_frag(sA, lc,  APITCH, kc, half);
        v16bf b = load_frag(sB, col, APITCH, kc, half);
        acc = wmma_bf16(a, b, acc);
    }
    float bias = bin[col];
#pragma unroll
    for (int i = 0; i < 8; ++i) {
        int row = i + 8 * half;                     // C layout: VGPR i -> M=i (+8 hi-half)
        h[(mt * 16 + row) * N_FILTERS + col] = acc[i] + bias;
    }
}

// -------------------------------------------------------------------------
// Kernel 2 (fused pair pipeline):
//   Wij = (ssp(f_ij@Wf1 + bf1) @ Wf2 + bf2) * rcut
//   atomicAdd(agg[idx_i], h[idx_j] * Wij)
// 5000 blocks x 8 pair-tiles of 16 pairs; weights staged once per block.
// The h[idx_j] gather (16 random 512B rows) is prefetched into LDS with
// global_load_async_to_lds_b128 *before* the GEMMs so its L2 latency is
// hidden behind the 5 WMMAs of the filter network.
// -------------------------------------------------------------------------
__global__ void pair_kernel(const float* __restrict__ f_ij,
                            const float* __restrict__ rcut,
                            const int*   __restrict__ idx_i,
                            const int*   __restrict__ idx_j,
                            const float* __restrict__ Wf1,
                            const float* __restrict__ bf1,
                            const float* __restrict__ Wf2,
                            const float* __restrict__ bf2,
                            const float* __restrict__ h,
                            float*       __restrict__ agg) {
    __shared__ __attribute__((aligned(16))) bf16 sW1[128 * FPITCH]; // col-major, K padded 20->32
    __shared__ __attribute__((aligned(16))) bf16 sW2[128 * APITCH]; // col-major
    __shared__ __attribute__((aligned(16))) bf16 sF[16 * FPITCH];
    __shared__ __attribute__((aligned(16))) bf16 sU[16 * APITCH];
    __shared__ __attribute__((aligned(16))) float sH[16 * N_FILTERS]; // gathered h rows
    __shared__ float sR[16];
    __shared__ int   sI[16], sJ[16];

    const int tid = threadIdx.x;
    for (int i = tid; i < 128 * 32; i += 256) {
        int n = i >> 5, k = i & 31;
        sW1[n * FPITCH + k] = (k < N_RBF) ? (bf16)Wf1[k * N_FILTERS + n] : (bf16)0.0f;
    }
    for (int i = tid; i < 128 * 128; i += 256) {
        int k = i >> 7, n = i & 127;
        sW2[n * APITCH + k] = (bf16)Wf2[k * N_FILTERS + n];
    }

    const int wave = tid >> 5, lane = tid & 31;
    const int half = lane >> 4, lc = lane & 15;
    const int col  = wave * 16 + lc;
    const float bias1 = bf1[col];
    const float bias2 = bf2[col];
    const unsigned sH_base = (unsigned)(size_t)(void*)sH;   // LDS byte offset of sH

    for (int t = 0; t < 8; ++t) {
        const int pbase = (blockIdx.x * 8 + t) * 16;
        __syncthreads();                            // protect sF/sU/sH (and weights on t=0)
        for (int i = tid; i < 16 * 32; i += 256) {
            int r = i >> 5, k = i & 31;
            sF[r * FPITCH + k] = (k < N_RBF) ? (bf16)f_ij[(pbase + r) * N_RBF + k]
                                             : (bf16)0.0f;
        }
        if (tid < 16) {
            sR[tid] = rcut[pbase + tid];
            sI[tid] = idx_i[pbase + tid];
            sJ[tid] = idx_j[pbase + tid];
        }
        // keep next tile's f_ij slab (1280B) flowing into the caches
        if (t < 7 && tid < 20)
            __builtin_prefetch(f_ij + (size_t)(pbase + 16) * N_RBF + tid * 16, 0, 0);
        __syncthreads();

        // Async gather: 16 rows x 512B of h -> sH.  512 b128 chunks, 2/thread.
#pragma unroll
        for (int c = 0; c < 2; ++c) {
            int id = tid + c * 256;                 // 0..511
            int r  = id >> 5;                       // gathered row 0..15
            int ob = (id & 31) << 4;                // byte offset within row
            const char* src = (const char*)(h + (size_t)sJ[r] * N_FILTERS) + ob;
            async_copy_b128(sH_base + (unsigned)(id << 4), src);
        }

        // GEMM1: [16x32pad] x [32pad x 16]  (one WMMA)
        {
            v16bf a = load_frag(sF, lc,  FPITCH, 0, half);
            v16bf b = load_frag(sW1, col, FPITCH, 0, half);
            v8f u = {};
            u = wmma_bf16(a, b, u);
#pragma unroll
            for (int i = 0; i < 8; ++i) {
                int row = i + 8 * half;
                sU[row * APITCH + col] = (bf16)ssp(u[i] + bias1);
            }
        }
        __syncthreads();

        // GEMM2: [16x128] x [128x16]  (4 WMMA)
        v8f acc = {};
#pragma unroll
        for (int kc = 0; kc < 128; kc += 32) {
            v16bf a = load_frag(sU, lc,  APITCH, kc, half);
            v16bf b = load_frag(sW2, col, APITCH, kc, half);
            acc = wmma_bf16(a, b, acc);
        }

        wait_async0();                              // own async transfers done
        __syncthreads();                            // everyone's transfers visible

        // cutoff * gathered-h multiply, then scatter-add into L2-resident agg
#pragma unroll
        for (int i = 0; i < 8; ++i) {
            int row = i + 8 * half;
            float wij = (acc[i] + bias2) * sR[row];
            float v   = sH[row * N_FILTERS + col] * wij;
            atomicAdd(&agg[sI[row] * N_FILTERS + col], v);
        }
    }
}

// -------------------------------------------------------------------------
// Kernel 3:  out = ssp(agg@Wo1 + bo1) @ Wo2 + bo2
// Wo1/Wo2 staged sequentially into one LDS buffer (stay < 64KB static LDS).
// -------------------------------------------------------------------------
__global__ void out_kernel(const float* __restrict__ agg,
                           const float* __restrict__ Wo1,
                           const float* __restrict__ bo1,
                           const float* __restrict__ Wo2,
                           const float* __restrict__ bo2,
                           float* __restrict__ out) {
    __shared__ __attribute__((aligned(16))) bf16 sW[128 * APITCH]; // col-major, reused
    __shared__ __attribute__((aligned(16))) bf16 sA[16 * APITCH];
    __shared__ __attribute__((aligned(16))) bf16 sT[16 * APITCH];
    const int tid = threadIdx.x;
    const int mt  = blockIdx.x;

    for (int i = tid; i < 16 * 128; i += 256) {
        int r = i >> 7, k = i & 127;
        sA[r * APITCH + k] = (bf16)agg[(mt * 16 + r) * N_FILTERS + k];
    }
    for (int i = tid; i < 128 * 128; i += 256) {
        int k = i >> 7, n = i & 127;
        sW[n * APITCH + k] = (bf16)Wo1[k * N_BASIS + n];
    }
    __syncthreads();

    const int wave = tid >> 5, lane = tid & 31;
    const int half = lane >> 4, lc = lane & 15;
    const int col  = wave * 16 + lc;

    v8f acc = {};
#pragma unroll
    for (int kc = 0; kc < 128; kc += 32) {
        v16bf a = load_frag(sA, lc,  APITCH, kc, half);
        v16bf b = load_frag(sW, col, APITCH, kc, half);
        acc = wmma_bf16(a, b, acc);
    }
    float bias1 = bo1[col];
#pragma unroll
    for (int i = 0; i < 8; ++i) {
        int row = i + 8 * half;
        sT[row * APITCH + col] = (bf16)ssp(acc[i] + bias1);
    }
    __syncthreads();
    for (int i = tid; i < 128 * 128; i += 256) {     // restage: Wo2 col-major
        int k = i >> 7, n = i & 127;
        sW[n * APITCH + k] = (bf16)Wo2[k * N_BASIS + n];
    }
    __syncthreads();

    v8f acc2 = {};
#pragma unroll
    for (int kc = 0; kc < 128; kc += 32) {
        v16bf a = load_frag(sT, lc,  APITCH, kc, half);
        v16bf b = load_frag(sW, col, APITCH, kc, half);
        acc2 = wmma_bf16(a, b, acc2);
    }
    float bias2 = bo2[col];
#pragma unroll
    for (int i = 0; i < 8; ++i) {
        int row = i + 8 * half;
        out[(mt * 16 + row) * N_BASIS + col] = acc2[i] + bias2;
    }
}

extern "C" void kernel_launch(void* const* d_in, const int* in_sizes, int n_in,
                              void* d_out, int out_size, void* d_ws, size_t ws_size,
                              hipStream_t stream) {
    const float* x     = (const float*)d_in[0];
    const float* f_ij  = (const float*)d_in[1];
    const float* rcut  = (const float*)d_in[2];
    const int*   idx_i = (const int*)  d_in[3];
    const int*   idx_j = (const int*)  d_in[4];
    const float* Win   = (const float*)d_in[5];
    const float* bin   = (const float*)d_in[6];
    const float* Wf1   = (const float*)d_in[7];
    const float* bf1   = (const float*)d_in[8];
    const float* Wf2   = (const float*)d_in[9];
    const float* bf2   = (const float*)d_in[10];
    const float* Wo1   = (const float*)d_in[11];
    const float* bo1   = (const float*)d_in[12];
    const float* Wo2   = (const float*)d_in[13];
    const float* bo2   = (const float*)d_in[14];
    float* out = (float*)d_out;

    float* h   = (float*)d_ws;                       // [20000,128]
    float* agg = h + (size_t)N_ATOMS * N_FILTERS;    // [20000,128]
    const int nAgg = N_ATOMS * N_FILTERS;

    zero_kernel<<<(nAgg + 255) / 256, 256, 0, stream>>>(agg, nAgg);
    gemm_h_kernel<<<N_ATOMS / 16, 256, 0, stream>>>(x, Win, bin, h);
    pair_kernel<<<N_PAIRS / (16 * 8), 256, 0, stream>>>(
        f_ij, rcut, idx_i, idx_j, Wf1, bf1, Wf2, bf2, h, agg);
    out_kernel<<<N_ATOMS / 16, 256, 0, stream>>>(agg, Wo1, bo1, Wo2, bo2, out);
}
